// MixtralBinaryDiff_5712306504044
// MI455X (gfx1250) — compile-verified, matrix operand order checked by reading, not run on hardware
//
#include <hip/hip_runtime.h>
#include <hip/hip_bf16.h>

typedef __bf16 bf16;
typedef __attribute__((ext_vector_type(16))) __bf16 v16bf;
typedef __attribute__((ext_vector_type(8)))  float  v8f;
typedef __attribute__((ext_vector_type(8)))  unsigned int v8u;
typedef int v4i_ __attribute__((vector_size(4 * sizeof(int))));

#define HID   4096
#define INTER 14336
#define TOK   4096   // B * S

#define AS1 __attribute__((address_space(1)))
#define AS3 __attribute__((address_space(3)))

#if defined(__gfx1250__) && __has_builtin(__builtin_amdgcn_global_load_async_to_lds_b128)
#define USE_ASYNC_LDS 1
#else
#define USE_ASYNC_LDS 0
#endif

// ---------------------------------------------------------------------------
// Staging helpers: 16-byte global -> LDS copy, async (ASYNCcnt) when available
// ---------------------------------------------------------------------------
__device__ __forceinline__ void stage16(bf16* dst_lds, const bf16* src_glb) {
#if USE_ASYNC_LDS
  __builtin_amdgcn_global_load_async_to_lds_b128(
      (AS1 v4i_*)src_glb, (AS3 v4i_*)dst_lds, 0, 0);
#else
  *(uint4*)dst_lds = *(const uint4*)src_glb;
#endif
}

__device__ __forceinline__ void async_wait_all() {
#if USE_ASYNC_LDS
#if __has_builtin(__builtin_amdgcn_s_wait_asynccnt)
  __builtin_amdgcn_s_wait_asynccnt(0);
#else
  asm volatile("s_wait_asynccnt 0" ::: "memory");
#endif
#endif
}

// ---------------------------------------------------------------------------
// Stage 1: per-block partial sums of |w - mean| (deterministic 2-stage reduce)
// ---------------------------------------------------------------------------
__global__ __launch_bounds__(256) void bd_reduce_absdiff(
    const float* __restrict__ w, const float* __restrict__ m,
    float* __restrict__ partial, long long n) {
  __shared__ float sm[256];
  long long i = (long long)blockIdx.x * blockDim.x + threadIdx.x;
  const long long stride = (long long)gridDim.x * blockDim.x;
  float s = 0.0f;
  for (; i < n; i += stride) s += fabsf(w[i] - m[i]);
  sm[threadIdx.x] = s;
  __syncthreads();
  for (int o = 128; o > 0; o >>= 1) {
    if ((int)threadIdx.x < o) sm[threadIdx.x] += sm[threadIdx.x + o];
    __syncthreads();
  }
  if (threadIdx.x == 0) partial[blockIdx.x] = sm[0];
}

// one block per weight matrix: reduce its 1024 partials -> coeff = sum / n
__global__ __launch_bounds__(256) void bd_finalize_coeff(
    const float* __restrict__ partial, float* __restrict__ coeff, float invN) {
  __shared__ float sm[256];
  const float* p = partial + (size_t)blockIdx.x * 1024;
  float s = p[threadIdx.x] + p[threadIdx.x + 256] +
            p[threadIdx.x + 512] + p[threadIdx.x + 768];
  sm[threadIdx.x] = s;
  __syncthreads();
  for (int o = 128; o > 0; o >>= 1) {
    if ((int)threadIdx.x < o) sm[threadIdx.x] += sm[threadIdx.x + o];
    __syncthreads();
  }
  if (threadIdx.x == 0) coeff[blockIdx.x] = sm[0] * invN;
}

// ---------------------------------------------------------------------------
// Stage 2: materialize bf16 binary-diff weight  Wb = bf16(mean + c*sign(w-mean))
// ---------------------------------------------------------------------------
__global__ __launch_bounds__(256) void bd_binarize_bf16(
    const float* __restrict__ w, const float* __restrict__ m,
    const float* __restrict__ coeff, int ci,
    bf16* __restrict__ out, long long n) {
  const float c = coeff[ci];
  long long i = (long long)blockIdx.x * blockDim.x + threadIdx.x;
  const long long stride = (long long)gridDim.x * blockDim.x;
  for (; i < n; i += stride) {
    float base = m[i];
    float d = w[i] - base;
    float sg = (d > 0.0f) ? 1.0f : ((d < 0.0f) ? -1.0f : 0.0f);
    out[i] = (bf16)(base + c * sg);
  }
}

__global__ __launch_bounds__(256) void bd_f32_to_bf16(
    const float* __restrict__ x, bf16* __restrict__ y, long long n) {
  long long i = (long long)blockIdx.x * blockDim.x + threadIdx.x;
  const long long stride = (long long)gridDim.x * blockDim.x;
  for (; i < n; i += stride) y[i] = (bf16)x[i];
}

// ---------------------------------------------------------------------------
// WMMA fragment gather from a 32-wide K slice in LDS
// (ISA 7.12.2, 16-bit operand layout; pairs are contiguous -> ds_load_b128 x2)
// ---------------------------------------------------------------------------
__device__ __forceinline__ v16bf load_frag16(const bf16* row, int laneHiOfs) {
  v8u u;
#pragma unroll
  for (int p = 0; p < 8; ++p) {
    const int kk = 2 * p + ((p < 4) ? 0 : 8) + laneHiOfs;  // even -> 4B aligned
    u[p] = *(const unsigned int*)(row + kk);
  }
  return __builtin_bit_cast(v16bf, u);
}

// ---------------------------------------------------------------------------
// GEMM 1: gate = Xb @ W1b^T, up = Xb @ W3b^T, Hb = silu(gate) * up   (bf16 out)
// Block 256 threads = 8 waves as 4(M) x 2(N); BM=128, BN=64, BK=32.
// Wave tile 32x32 for both matrices -> 8 v_wmma per K-step per wave.
// Double-buffered LDS: async loads of tile t+1 overlap WMMAs on tile t.
// ---------------------------------------------------------------------------
__global__ __launch_bounds__(256) void bd_ffn_gate_up(
    const bf16* __restrict__ Xb, const bf16* __restrict__ W1b,
    const bf16* __restrict__ W3b, bf16* __restrict__ Hb) {
  __shared__ __align__(16) bf16 Xs[2][128][32];   // 16 KB
  __shared__ __align__(16) bf16 W1s[2][64][32];   //  8 KB
  __shared__ __align__(16) bf16 W3s[2][64][32];   //  8 KB

  const int tid    = threadIdx.x;
  const int lane   = tid & 31;
  const int wave   = tid >> 5;
  const int laneLo = lane & 15;
  const int laneHiOfs = (lane >> 4) ? 8 : 0;
  const int mwave = wave & 3;   // 0..3 -> 32-row M subtile
  const int nwave = wave >> 2;  // 0..1 -> 32-col N subtile

  const int m0 = blockIdx.y * 128;
  const int n0 = blockIdx.x * 64;

  // staging coordinates (16B = 8 bf16 chunks)
  const int xr0 = tid >> 1;               // Xs row (2 chunks/thread)
  const int xc0 = (tid & 1) * 16;
  const int wr  = tid >> 2;               // W row (1 chunk/thread)
  const int wc  = (tid & 3) * 8;

  const bf16* xsrc = Xb  + (size_t)(m0 + xr0) * HID + xc0;
  const bf16* w1src = W1b + (size_t)(n0 + wr) * HID + wc;
  const bf16* w3src = W3b + (size_t)(n0 + wr) * HID + wc;

  const v8f vzero = {0.f, 0.f, 0.f, 0.f, 0.f, 0.f, 0.f, 0.f};
  v8f accG[2][2] = {{vzero, vzero}, {vzero, vzero}};
  v8f accU[2][2] = {{vzero, vzero}, {vzero, vzero}};

  const int KT = HID / 32;

  // prologue: stage tile 0 into buffer 0
  stage16(&Xs[0][xr0][xc0],     xsrc);
  stage16(&Xs[0][xr0][xc0 + 8], xsrc + 8);
  stage16(&W1s[0][wr][wc],      w1src);
  stage16(&W3s[0][wr][wc],      w3src);
  async_wait_all();
  __syncthreads();

  for (int t = 0; t < KT; ++t) {
    const int cur = t & 1;
    const int nxt = cur ^ 1;
    if (t + 1 < KT) {
      const int k1 = (t + 1) * 32;
      stage16(&Xs[nxt][xr0][xc0],     xsrc + k1);
      stage16(&Xs[nxt][xr0][xc0 + 8], xsrc + k1 + 8);
      stage16(&W1s[nxt][wr][wc],      w1src + k1);
      stage16(&W3s[nxt][wr][wc],      w3src + k1);
    }

    v16bf a[2], b1[2], b3[2];
#pragma unroll
    for (int mi = 0; mi < 2; ++mi)
      a[mi] = load_frag16(&Xs[cur][mwave * 32 + mi * 16 + laneLo][0], laneHiOfs);
#pragma unroll
    for (int ni = 0; ni < 2; ++ni) {
      b1[ni] = load_frag16(&W1s[cur][nwave * 32 + ni * 16 + laneLo][0], laneHiOfs);
      b3[ni] = load_frag16(&W3s[cur][nwave * 32 + ni * 16 + laneLo][0], laneHiOfs);
    }
#pragma unroll
    for (int mi = 0; mi < 2; ++mi) {
#pragma unroll
      for (int ni = 0; ni < 2; ++ni) {
        accG[mi][ni] = __builtin_amdgcn_wmma_f32_16x16x32_bf16(
            false, a[mi], false, b1[ni], (short)0, accG[mi][ni], false, false);
        accU[mi][ni] = __builtin_amdgcn_wmma_f32_16x16x32_bf16(
            false, a[mi], false, b3[ni], (short)0, accU[mi][ni], false, false);
      }
    }

    if (t + 1 < KT) {
      async_wait_all();   // tile t+1 resident
      __syncthreads();    // and everyone done reading buffer `cur`
    }
  }

  // epilogue: h = silu(g) * u ; C/D layout: VGPR r -> M = r (+8 for hi lanes)
  const int rBase = m0 + mwave * 32 + (laneHiOfs ? 8 : 0);
#pragma unroll
  for (int mi = 0; mi < 2; ++mi) {
#pragma unroll
    for (int ni = 0; ni < 2; ++ni) {
      const int col = n0 + nwave * 32 + ni * 16 + laneLo;
#pragma unroll
      for (int r = 0; r < 8; ++r) {
        const float g = accG[mi][ni][r];
        const float u = accU[mi][ni][r];
        const float h = g * __builtin_amdgcn_rcpf(1.0f + __expf(-g)) * u;
        Hb[(size_t)(rBase + mi * 16 + r) * INTER + col] = (bf16)h;
      }
    }
  }
}

// ---------------------------------------------------------------------------
// GEMM 2: out = Hb @ W2b^T   (f32 out, K = INTER)
// BM=128, BN=128, BK=32; 8 waves as 4(M) x 2(N); wave tile 32x64
// -> 2x4 accumulators, 8 v_wmma per K-step per wave. Double-buffered LDS.
// ---------------------------------------------------------------------------
__global__ __launch_bounds__(256) void bd_ffn_down(
    const bf16* __restrict__ Hb, const bf16* __restrict__ W2b,
    float* __restrict__ Out) {
  __shared__ __align__(16) bf16 Hs[2][128][32];   // 16 KB
  __shared__ __align__(16) bf16 Ws[2][128][32];   // 16 KB

  const int tid    = threadIdx.x;
  const int lane   = tid & 31;
  const int wave   = tid >> 5;
  const int laneLo = lane & 15;
  const int laneHiOfs = (lane >> 4) ? 8 : 0;
  const int mwave = wave & 3;   // 32-row M subtile
  const int nwave = wave >> 2;  // 64-col N subtile

  const int m0 = blockIdx.y * 128;
  const int n0 = blockIdx.x * 128;

  const int xr0 = tid >> 1;               // 2 chunks/thread for each buffer
  const int xc0 = (tid & 1) * 16;

  const bf16* hsrc = Hb  + (size_t)(m0 + xr0) * INTER + xc0;
  const bf16* wsrc = W2b + (size_t)(n0 + xr0) * INTER + xc0;

  const v8f vzero = {0.f, 0.f, 0.f, 0.f, 0.f, 0.f, 0.f, 0.f};
  v8f acc[2][4] = {{vzero, vzero, vzero, vzero}, {vzero, vzero, vzero, vzero}};

  const int KT = INTER / 32;

  stage16(&Hs[0][xr0][xc0],     hsrc);
  stage16(&Hs[0][xr0][xc0 + 8], hsrc + 8);
  stage16(&Ws[0][xr0][xc0],     wsrc);
  stage16(&Ws[0][xr0][xc0 + 8], wsrc + 8);
  async_wait_all();
  __syncthreads();

  for (int t = 0; t < KT; ++t) {
    const int cur = t & 1;
    const int nxt = cur ^ 1;
    if (t + 1 < KT) {
      const int k1 = (t + 1) * 32;
      stage16(&Hs[nxt][xr0][xc0],     hsrc + k1);
      stage16(&Hs[nxt][xr0][xc0 + 8], hsrc + k1 + 8);
      stage16(&Ws[nxt][xr0][xc0],     wsrc + k1);
      stage16(&Ws[nxt][xr0][xc0 + 8], wsrc + k1 + 8);
    }

    v16bf a[2], b[4];
#pragma unroll
    for (int mi = 0; mi < 2; ++mi)
      a[mi] = load_frag16(&Hs[cur][mwave * 32 + mi * 16 + laneLo][0], laneHiOfs);
#pragma unroll
    for (int ni = 0; ni < 4; ++ni)
      b[ni] = load_frag16(&Ws[cur][nwave * 64 + ni * 16 + laneLo][0], laneHiOfs);
#pragma unroll
    for (int mi = 0; mi < 2; ++mi)
#pragma unroll
      for (int ni = 0; ni < 4; ++ni)
        acc[mi][ni] = __builtin_amdgcn_wmma_f32_16x16x32_bf16(
            false, a[mi], false, b[ni], (short)0, acc[mi][ni], false, false);

    if (t + 1 < KT) {
      async_wait_all();
      __syncthreads();
    }
  }

  const int rBase = m0 + mwave * 32 + (laneHiOfs ? 8 : 0);
#pragma unroll
  for (int mi = 0; mi < 2; ++mi) {
#pragma unroll
    for (int ni = 0; ni < 4; ++ni) {
      const int col = n0 + nwave * 64 + ni * 16 + laneLo;
#pragma unroll
      for (int r = 0; r < 8; ++r) {
        Out[(size_t)(rBase + mi * 16 + r) * HID + col] = acc[mi][ni][r];
      }
    }
  }
}

// ---------------------------------------------------------------------------
// Launcher
// ---------------------------------------------------------------------------
extern "C" void kernel_launch(void* const* d_in, const int* in_sizes, int n_in,
                              void* d_out, int out_size, void* d_ws, size_t ws_size,
                              hipStream_t stream) {
  (void)in_sizes; (void)n_in; (void)out_size; (void)ws_size;

  const float* x   = (const float*)d_in[0];  // [TOK, HID]
  const float* w1  = (const float*)d_in[1];  // [INTER, HID]
  const float* mw1 = (const float*)d_in[2];
  const float* w2  = (const float*)d_in[3];  // [HID, INTER]
  const float* mw2 = (const float*)d_in[4];
  const float* w3  = (const float*)d_in[5];  // [INTER, HID]
  const float* mw3 = (const float*)d_in[6];
  float* out = (float*)d_out;                // [TOK, HID]

  const long long NW = (long long)INTER * HID;  // 58,720,256
  const long long NX = (long long)TOK * HID;    // 16,777,216

  char* ws = (char*)d_ws;
  float* partials = (float*)(ws);                 // 3 * 1024 floats
  float* coeff    = (float*)(ws + 16384);         // 3 floats
  size_t off = 65536;
  bf16* Xb  = (bf16*)(ws + off); off += (size_t)NX * sizeof(bf16);
  bf16* W1b = (bf16*)(ws + off); off += (size_t)NW * sizeof(bf16);
  bf16* W3b = (bf16*)(ws + off); off += (size_t)NW * sizeof(bf16);
  bf16* W2b = (bf16*)(ws + off); off += (size_t)NW * sizeof(bf16);
  bf16* Hb  = (bf16*)(ws + off);

  // coeff reductions (w1 -> 0, w2 -> 1, w3 -> 2)
  bd_reduce_absdiff<<<1024, 256, 0, stream>>>(w1, mw1, partials + 0 * 1024, NW);
  bd_reduce_absdiff<<<1024, 256, 0, stream>>>(w2, mw2, partials + 1 * 1024, NW);
  bd_reduce_absdiff<<<1024, 256, 0, stream>>>(w3, mw3, partials + 2 * 1024, NW);
  bd_finalize_coeff<<<3, 256, 0, stream>>>(partials, coeff, 1.0f / (float)NW);

  // materialize bf16 weights + activations
  bd_binarize_bf16<<<8192, 256, 0, stream>>>(w1, mw1, coeff, 0, W1b, NW);
  bd_binarize_bf16<<<8192, 256, 0, stream>>>(w2, mw2, coeff, 1, W2b, NW);
  bd_binarize_bf16<<<8192, 256, 0, stream>>>(w3, mw3, coeff, 2, W3b, NW);
  bd_f32_to_bf16<<<8192, 256, 0, stream>>>(x, Xb, NX);

  // fused gate/up GEMM + SiLU, then down GEMM
  bd_ffn_gate_up<<<dim3(INTER / 64, TOK / 128), 256, 0, stream>>>(Xb, W1b, W3b, Hb);
  bd_ffn_down<<<dim3(HID / 128, TOK / 128), 256, 0, stream>>>(Hb, W2b, out);
}
